// GPT2Attention_71305047048847
// MI455X (gfx1250) — matrix operand order repaired and checked by reading, not compiled, not run on hardware
//
#include <hip/hip_runtime.h>

#define SEQ    2048
#define DMODEL 1024
#define NH     16
#define HS     64

typedef __attribute__((ext_vector_type(16))) __bf16       v16bf;
typedef __attribute__((ext_vector_type(8)))  float        v8f;
typedef __attribute__((ext_vector_type(4)))  unsigned int v4u;

union V16U { v16bf v; v4u q[2]; };

__device__ __forceinline__ unsigned short f2bf(float x) {
    unsigned int u = __builtin_bit_cast(unsigned int, x);
    u = (u + 0x7FFFu + ((u >> 16) & 1u)) >> 16;   // round-to-nearest-even
    return (unsigned short)u;
}

__device__ __forceinline__ v8f vzero8() {
    v8f z;
#pragma unroll
    for (int i = 0; i < 8; ++i) z[i] = 0.0f;
    return z;
}

// ---------------- fp32 -> bf16 elementwise ----------------
__global__ void k_conv(const float* __restrict__ src, unsigned short* __restrict__ dst,
                       long long n) {
    long long i = (long long)blockIdx.x * blockDim.x + threadIdx.x;
    long long stride = (long long)gridDim.x * blockDim.x;
    for (; i < n; i += stride) dst[i] = f2bf(src[i]);
}

// ---------------- fp32 [K,N] -> bf16 [N,K] transpose ----------------
__global__ void k_convT(const float* __restrict__ src, unsigned short* __restrict__ dst,
                        int K, int N) {
    long long tot = (long long)K * N;
    long long i = (long long)blockIdx.x * blockDim.x + threadIdx.x;
    long long stride = (long long)gridDim.x * blockDim.x;
    for (; i < tot; i += stride) {
        int nn = (int)(i / K), kk = (int)(i % K);
        dst[i] = f2bf(src[(long long)kk * N + nn]);
    }
}

// =====================================================================
// GEMM core: wave tile 32(M) x 64(N) = 2x4 WMMA accumulators.
// Block = 8 waves arranged 2(M) x 4(N) -> block tile 64M x 256N.
// A[M,K] bf16 row-major; Bt[N,K] bf16 (pre-transposed weights).
// =====================================================================
#define GEMM_CORE(A, Bt, K)                                                              \
    const int lane = threadIdx.x & 31;                                                   \
    const int wave = threadIdx.x >> 5;                                                   \
    const int wm = wave >> 2, wn = wave & 3;                                             \
    const int m0 = blockIdx.y * 64 + wm * 32;                                            \
    const int n0 = blockIdx.x * 256 + wn * 64;                                           \
    const int ln = lane & 15, hv = lane >> 4;                                            \
    v8f acc[2][4];                                                                       \
    _Pragma("unroll") for (int mt = 0; mt < 2; ++mt)                                     \
        _Pragma("unroll") for (int nt = 0; nt < 4; ++nt) acc[mt][nt] = vzero8();         \
    const unsigned short* ap[2];                                                         \
    ap[0] = A + (size_t)(m0 + ln) * K;                                                   \
    ap[1] = A + (size_t)(m0 + 16 + ln) * K;                                              \
    const unsigned short* bp[4];                                                         \
    _Pragma("unroll") for (int nt = 0; nt < 4; ++nt)                                     \
        bp[nt] = Bt + (size_t)(n0 + nt * 16 + ln) * K + hv * 8;                          \
    for (int k0 = 0; k0 < K; k0 += 32) {                                                 \
        V16U af[2], bf[4];                                                               \
        _Pragma("unroll") for (int mt = 0; mt < 2; ++mt) {                               \
            af[mt].q[0] = *(const v4u*)(ap[mt] + k0 + hv * 8);                           \
            af[mt].q[1] = *(const v4u*)(ap[mt] + k0 + 16 + hv * 8);                      \
        }                                                                                \
        _Pragma("unroll") for (int nt = 0; nt < 4; ++nt) {                               \
            bf[nt].q[0] = *(const v4u*)(bp[nt] + k0);                                    \
            bf[nt].q[1] = *(const v4u*)(bp[nt] + k0 + 16);                               \
        }                                                                                \
        _Pragma("unroll") for (int mt = 0; mt < 2; ++mt)                                 \
            _Pragma("unroll") for (int nt = 0; nt < 4; ++nt)                             \
                acc[mt][nt] = __builtin_amdgcn_wmma_f32_16x16x32_bf16(                   \
                    false, af[mt].v, false, bf[nt].v, (short)0, acc[mt][nt], false, false); \
    }

// QKV GEMM: epilogue splits heads: Q,K -> [B*H,S,64]; V -> transposed [B*H,64,S]
__global__ __launch_bounds__(256) void k_gemm_qkv(
    const unsigned short* __restrict__ A,
    const unsigned short* __restrict__ Bt,
    const float* __restrict__ bias,
    unsigned short* __restrict__ Qo,
    unsigned short* __restrict__ Ko,
    unsigned short* __restrict__ Vt) {
    GEMM_CORE(A, Bt, DMODEL)
    const int sec = (blockIdx.x * 256) / DMODEL;   // 0=Q,1=K,2=V (256-block is section-pure)
#pragma unroll
    for (int mt = 0; mt < 2; ++mt)
#pragma unroll
        for (int nt = 0; nt < 4; ++nt) {
            const int colg = n0 + nt * 16 + ln;
            const int hh = (colg % DMODEL) >> 6;   // head (16-col tile is head-pure)
            const int c  = colg & 63;              // channel within head
            const float bs = bias[colg];
#pragma unroll
            for (int r = 0; r < 8; ++r) {
                const int row = m0 + mt * 16 + hv * 8 + r;
                const int bb  = row >> 11;         // / SEQ
                const int ss  = row & (SEQ - 1);
                const unsigned short val = f2bf(acc[mt][nt][r] + bs);
                const size_t bhh = (size_t)(bb * NH + hh);
                if (sec == 0)      Qo[(bhh * SEQ + ss) * HS + c] = val;
                else if (sec == 1) Ko[(bhh * SEQ + ss) * HS + c] = val;
                else               Vt[(bhh * HS + c) * SEQ + ss] = val;
            }
        }
}

// Output projection: ctx[M,1024] x Wt[1024,1024] + bias -> f32 out
__global__ __launch_bounds__(256) void k_gemm_proj(
    const unsigned short* __restrict__ A,
    const unsigned short* __restrict__ Bt,
    const float* __restrict__ bias,
    float* __restrict__ Out) {
    GEMM_CORE(A, Bt, DMODEL)
#pragma unroll
    for (int mt = 0; mt < 2; ++mt)
#pragma unroll
        for (int nt = 0; nt < 4; ++nt) {
            const int colg = n0 + nt * 16 + ln;
            const float bs = bias[colg];
#pragma unroll
            for (int r = 0; r < 8; ++r) {
                const int row = m0 + mt * 16 + hv * 8 + r;
                Out[(size_t)row * DMODEL + colg] = acc[mt][nt][r] + bs;
            }
        }
}

// =====================================================================
// Flash attention. Block = 8 waves, 16 queries each (128/block).
// K/V tiles double-buffered into LDS with global_load_async_to_lds_b128,
// overlapped with WMMA + online softmax; s_wait_asynccnt synchronizes.
// =====================================================================
__global__ __launch_bounds__(256) void k_attn(
    const unsigned short* __restrict__ Q,
    const unsigned short* __restrict__ Kx,
    const unsigned short* __restrict__ Vt,
    unsigned short* __restrict__ CTX) {
    __shared__ unsigned short Klds[2][32 * 64];   // [buf][key 0..31][c 0..63]
    __shared__ unsigned short Vlds[2][64 * 32];   // [buf][c 0..63][key 0..31]
    __shared__ unsigned short Plds[8][16 * 32];   // per-wave private P scratch

    const int tid  = threadIdx.x;
    const int lane = tid & 31;
    const int wave = tid >> 5;
    const int ln = lane & 15, hv = lane >> 4;
    const int bh = blockIdx.y;
    const int bb = bh >> 4, hh = bh & 15;
    const int q0 = blockIdx.x * 128;
    const int qw = q0 + wave * 16;

    const unsigned short* Qb = Q  + ((size_t)bh * SEQ + qw) * HS;
    const unsigned short* Kb = Kx + (size_t)bh * SEQ * HS;
    const unsigned short* Vb = Vt + (size_t)bh * HS * SEQ;

    // Q fragments, held for all key tiles
    V16U qa[2];
#pragma unroll
    for (int ch = 0; ch < 2; ++ch) {
        const unsigned short* qr = Qb + (size_t)ln * HS + ch * 32;
        qa[ch].q[0] = *(const v4u*)(qr + hv * 8);
        qa[ch].q[1] = *(const v4u*)(qr + 16 + hv * 8);
    }

    v8f ctx[4];
#pragma unroll
    for (int cc = 0; cc < 4; ++cc) ctx[cc] = vzero8();
    float Mx[8], Lsum[8];
#pragma unroll
    for (int r = 0; r < 8; ++r) { Mx[r] = -1e30f; Lsum[r] = 0.0f; }

    // async stage of key-tile kt into buffer buf: 1x b128 per thread per tile
    auto issue_tile = [&](int kt, int buf) {
        const int kb = kt * 32;
        {   // K tile: 32 rows x 64 halves (128B/row, 8 chunks)
            const unsigned short* gp = Kb + (size_t)(kb + (tid >> 3)) * HS + (tid & 7) * 8;
            unsigned lds_off = (unsigned)(size_t)(&Klds[buf][(size_t)tid * 8]);
            asm volatile("global_load_async_to_lds_b128 %0, %1, off"
                         :: "v"(lds_off), "v"((unsigned long long)(size_t)gp) : "memory");
        }
        {   // V tile: 64 c-rows x 32 halves (64B/row, 4 chunks)
            const unsigned short* gp = Vb + (size_t)(tid >> 2) * SEQ + kb + (tid & 3) * 8;
            unsigned lds_off = (unsigned)(size_t)(&Vlds[buf][(size_t)tid * 8]);
            asm volatile("global_load_async_to_lds_b128 %0, %1, off"
                         :: "v"(lds_off), "v"((unsigned long long)(size_t)gp) : "memory");
        }
    };

    const int nkt = (q0 >> 5) + 4;                // causal: keys up to q0+127
    issue_tile(0, 0);

    for (int kt = 0; kt < nkt; ++kt) {
        const int kb  = kt * 32;
        const int buf = kt & 1;

        __syncthreads();                          // prior reads of buf^1 finished
        if (kt + 1 < nkt) {
            issue_tile(kt + 1, buf ^ 1);
            asm volatile("s_wait_asynccnt 0x2" ::: "memory");  // current tile landed
        } else {
            asm volatile("s_wait_asynccnt 0x0" ::: "memory");
        }
        __syncthreads();                          // all waves' async data visible

        // ---- scores S[16q x 32k] = Q . K^T (K from LDS) ----
        v8f s0 = vzero8(), s1 = vzero8();
#pragma unroll
        for (int ch = 0; ch < 2; ++ch) {
            V16U k0f, k1f;
            const unsigned short* kr0 = &Klds[buf][(size_t)(ln) * 64      + ch * 32 + hv * 8];
            const unsigned short* kr1 = &Klds[buf][(size_t)(16 + ln) * 64 + ch * 32 + hv * 8];
            k0f.q[0] = *(const v4u*)(kr0);  k0f.q[1] = *(const v4u*)(kr0 + 16);
            k1f.q[0] = *(const v4u*)(kr1);  k1f.q[1] = *(const v4u*)(kr1 + 16);
            s0 = __builtin_amdgcn_wmma_f32_16x16x32_bf16(false, qa[ch].v, false, k0f.v, (short)0, s0, false, false);
            s1 = __builtin_amdgcn_wmma_f32_16x16x32_bf16(false, qa[ch].v, false, k1f.v, (short)0, s1, false, false);
        }

        // ---- scale, causal mask, online softmax ----
        float p0[8], p1[8];
#pragma unroll
        for (int r = 0; r < 8; ++r) {
            const int row = qw + hv * 8 + r;
            const int c0 = kb + ln, c1 = kb + 16 + ln;
            const float v0 = (c0 <= row) ? s0[r] * 0.125f : -10000.0f;
            const float v1 = (c1 <= row) ? s1[r] * 0.125f : -10000.0f;
            float t = fmaxf(v0, v1);
#pragma unroll
            for (int mk = 1; mk < 16; mk <<= 1) t = fmaxf(t, __shfl_xor(t, mk, 32));
            const float mnew = fmaxf(Mx[r], t);
            p0[r] = __expf(v0 - mnew);
            p1[r] = __expf(v1 - mnew);
            float sm = p0[r] + p1[r];
#pragma unroll
            for (int mk = 1; mk < 16; mk <<= 1) sm += __shfl_xor(sm, mk, 32);
            const float alpha = __expf(Mx[r] - mnew);
            Lsum[r] = Lsum[r] * alpha + sm;
            Mx[r] = mnew;
#pragma unroll
            for (int cc = 0; cc < 4; ++cc) ctx[cc][r] *= alpha;
        }

        // ---- P: C-layout regs -> LDS -> A-layout fragment ----
#pragma unroll
        for (int r = 0; r < 8; ++r) {
            const int row = hv * 8 + r;
            Plds[wave][row * 32 + ln]      = f2bf(p0[r]);
            Plds[wave][row * 32 + 16 + ln] = f2bf(p1[r]);
        }
        __builtin_amdgcn_wave_barrier();          // keep DS store -> DS load order
        V16U pa;
        {
            const unsigned short* pr = &Plds[wave][ln * 32];
            pa.q[0] = *(const v4u*)(pr + hv * 8);
            pa.q[1] = *(const v4u*)(pr + 16 + hv * 8);
        }
        __builtin_amdgcn_wave_barrier();

        // ---- ctx += P . V  (V from LDS, transposed [c][k]) ----
#pragma unroll
        for (int cc = 0; cc < 4; ++cc) {
            V16U vf;
            const unsigned short* vr = &Vlds[buf][(size_t)(cc * 16 + ln) * 32 + hv * 8];
            vf.q[0] = *(const v4u*)(vr);
            vf.q[1] = *(const v4u*)(vr + 16);
            ctx[cc] = __builtin_amdgcn_wmma_f32_16x16x32_bf16(false, pa.v, false, vf.v, (short)0, ctx[cc], false, false);
        }
    }

    // ---- normalize + write ctx back token-major [B, S, D] (bf16) ----
#pragma unroll
    for (int cc = 0; cc < 4; ++cc)
#pragma unroll
        for (int r = 0; r < 8; ++r) {
            const int ss = qw + hv * 8 + r;
            const float v = ctx[cc][r] / Lsum[r];
            CTX[((size_t)bb * SEQ + ss) * DMODEL + hh * HS + cc * 16 + ln] = f2bf(v);
        }
}

// =====================================================================
static inline int gblocks(long long n, int t) {
    long long b = (n + t - 1) / t;
    if (b > 32768) b = 32768;
    return (int)b;
}

extern "C" void kernel_launch(void* const* d_in, const int* in_sizes, int n_in,
                              void* d_out, int out_size, void* d_ws, size_t ws_size,
                              hipStream_t stream) {
    const float* enc    = (const float*)d_in[0];
    const float* w_attn = (const float*)d_in[1];
    const float* b_attn = (const float*)d_in[2];
    const float* w_proj = (const float*)d_in[3];
    const float* b_proj = (const float*)d_in[4];
    float* out = (float*)d_out;

    const int Bsz   = in_sizes[0] / (SEQ * DMODEL);
    const int Mrows = Bsz * SEQ;

    // workspace carve (bf16 halves)
    unsigned short* Xb  = (unsigned short*)d_ws;
    unsigned short* Wat = Xb  + (size_t)Mrows * DMODEL;      // [3072,1024]
    unsigned short* Wpt = Wat + (size_t)3 * DMODEL * DMODEL; // [1024,1024]
    unsigned short* Qw  = Wpt + (size_t)DMODEL * DMODEL;     // [B*H,S,64]
    unsigned short* Kw  = Qw  + (size_t)Mrows * DMODEL;
    unsigned short* Vw  = Kw  + (size_t)Mrows * DMODEL;      // [B*H,64,S]
    unsigned short* Cw  = Vw  + (size_t)Mrows * DMODEL;      // [B,S,D]

    // 1) precision conversion (+ weight transposes)
    long long nX = (long long)Mrows * DMODEL;
    k_conv<<<gblocks(nX, 256), 256, 0, stream>>>(enc, Xb, nX);
    k_convT<<<gblocks((long long)3 * DMODEL * DMODEL, 256), 256, 0, stream>>>(w_attn, Wat, DMODEL, 3 * DMODEL);
    k_convT<<<gblocks((long long)DMODEL * DMODEL, 256), 256, 0, stream>>>(w_proj, Wpt, DMODEL, DMODEL);

    // 2) QKV projection (head-split epilogue, V transposed)
    dim3 g1(3 * DMODEL / 256, Mrows / 64);
    k_gemm_qkv<<<g1, 256, 0, stream>>>(Xb, Wat, b_attn, Qw, Kw, Vw);

    // 3) causal flash attention (async LDS staging of K/V)
    dim3 g2(SEQ / 128, Bsz * NH);
    k_attn<<<g2, 256, 0, stream>>>(Qw, Kw, Vw, Cw);

    // 4) output projection -> fp32 d_out
    dim3 g3(DMODEL / 256, Mrows / 64);
    k_gemm_proj<<<g3, 256, 0, stream>>>(Cw, Wpt, b_proj, out);
}
